// LSTMModel_22960895165009
// MI455X (gfx1250) — matrix operand order repaired and checked by reading, not compiled, not run on hardware
//
#include <hip/hip_runtime.h>
#include <hip/hip_bf16.h>

typedef __attribute__((ext_vector_type(16))) _Float16 v16h;
typedef __attribute__((ext_vector_type(8)))  _Float16 v8h;
typedef __attribute__((ext_vector_type(8)))  float    v8f;

#define WMMA_F16(a, b, c) \
  __builtin_amdgcn_wmma_f32_16x16x32_f16(false, (a), false, (b), (short)0, (c), false, false)

static __device__ __forceinline__ v16h cat8(v8h lo, v8h hi) {
  return __builtin_shufflevector(lo, hi, 0,1,2,3,4,5,6,7,8,9,10,11,12,13,14,15);
}

// ---------------------------------------------------------------------------
// fp32 -> f16 with K padding (row-major rows x K -> rows x Kp, zero pad)
// ---------------------------------------------------------------------------
__global__ __launch_bounds__(256)
void k_cvt_pad(const float* __restrict__ src, _Float16* __restrict__ dst,
               int rows, int K, int Kp) {
  long i = blockIdx.x * 256L + threadIdx.x;
  long total = (long)rows * Kp;
  if (i >= total) return;
  int  col = (int)(i % Kp);
  long row = i / Kp;
  dst[i] = (col < K) ? (_Float16)src[row * (long)K + col] : (_Float16)0.f;
}

// ---------------------------------------------------------------------------
// transpose weight (K x N fp32) -> (N x Kp f16), zero pad K
// ---------------------------------------------------------------------------
__global__ __launch_bounds__(256)
void k_transpose_pad(const float* __restrict__ W, _Float16* __restrict__ Wt,
                     int K, int N, int Kp) {
  long i = blockIdx.x * 256L + threadIdx.x;
  long total = (long)N * Kp;
  if (i >= total) return;
  int k = (int)(i % Kp);
  int n = (int)(i / Kp);
  Wt[i] = (k < K) ? (_Float16)W[(long)k * N + n] : (_Float16)0.f;
}

// ---------------------------------------------------------------------------
// Z[M x N] = A[M x Kp] (f16) @ Wt[N x Kp]^T (f16) + bias (f32), store f16
// grid: (M/64, N/128), block 256 (8 waves); wave = 16x64 tile
// Fragment layouts use contiguous k = 16*hi + s in A and B (16-bit A: M=lane&15, K in VGPRs;
// B via pre-transposed weight so lane&15 = N, 16 contiguous K halves).
// ---------------------------------------------------------------------------
__global__ __launch_bounds__(256)
void k_gemm_bias(const _Float16* __restrict__ A, const _Float16* __restrict__ Wt,
                 const float* __restrict__ bias, _Float16* __restrict__ Z,
                 int Kp, int N) {
  const int lane = threadIdx.x & 31;
  const int w    = threadIdx.x >> 5;
  const int wm   = w & 3;
  const int wn   = w >> 2;
  const int lm   = lane & 15;
  const int hi   = lane >> 4;
  const int m0   = blockIdx.x * 64 + wm * 16;
  const int n0   = blockIdx.y * 128 + wn * 64;

  v8f acc[4] = {};
  const _Float16* Arow = A + (long)(m0 + lm) * Kp;
  for (int k0 = 0; k0 < Kp; k0 += 32) {
    v8h a_lo = *(const v8h*)(Arow + k0 + hi * 16);
    v8h a_hi = *(const v8h*)(Arow + k0 + hi * 16 + 8);
    v16h a = cat8(a_lo, a_hi);
#pragma unroll
    for (int nt = 0; nt < 4; ++nt) {
      const _Float16* Brow = Wt + (long)(n0 + nt * 16 + lm) * Kp + k0 + hi * 16;
      v16h b = *(const v16h*)Brow;
      acc[nt] = WMMA_F16(a, b, acc[nt]);
    }
  }
#pragma unroll
  for (int nt = 0; nt < 4; ++nt) {
    int col = n0 + nt * 16 + lm;
    float bv = bias[col];
#pragma unroll
    for (int r = 0; r < 8; ++r) {
      int row = m0 + r + hi * 8;
      Z[(long)row * N + col] = (_Float16)(acc[nt][r] + bv);
    }
  }
}

// ---------------------------------------------------------------------------
// Recurrent LSTM scan. One workgroup = 16 batch samples for all T steps.
// block = 32 * (U/16) threads; grid = B/16. Wave w owns units [w*16, w*16+16)
// and computes ALL FOUR gate accumulators for that slice, so the cell update
// is entirely in-register. h lives in LDS (f16) and is rebuilt each step
// between two barriers; c lives in VGPRs.
// XZ: (B*T) x 4U f16 (precomputed x@W + b, gate order i,f,g,o)
// Ut: (4U) x U f16  (transposed recurrent weights)
// ---------------------------------------------------------------------------
template<int U>
__global__ __launch_bounds__(256)
void k_lstm_rec(const _Float16* __restrict__ XZ, const _Float16* __restrict__ Ut,
                _Float16* __restrict__ Hseq, int T) {
  constexpr int LSTR = U + 8;           // padded LDS row stride (halves)
  __shared__ _Float16 hbuf[16 * LSTR];

  const int lane = threadIdx.x & 31;
  const int w    = threadIdx.x >> 5;    // n-tile index, 0..U/16-1
  const int lm   = lane & 15;
  const int hi   = lane >> 4;
  const int b0   = blockIdx.x * 16;

  for (int i = threadIdx.x; i < 16 * LSTR; i += blockDim.x) hbuf[i] = (_Float16)0.f;
  __syncthreads();

  float c[8];
#pragma unroll
  for (int r = 0; r < 8; ++r) c[r] = 0.f;

  const int ncol = w * 16 + lm;         // unit index within u

  for (int t = 0; t < T; ++t) {
    v8f acc[4] = {};                    // gates i,f,g,o
#pragma unroll
    for (int k0 = 0; k0 < U; k0 += 32) {
      const _Float16* ar = &hbuf[lm * LSTR + k0 + hi * 16];
      v16h a = cat8(*(const v8h*)ar, *(const v8h*)(ar + 8));
#pragma unroll
      for (int g = 0; g < 4; ++g) {
        const _Float16* br = Ut + (long)(g * U + ncol) * U + k0 + hi * 16;
        v16h b = *(const v16h*)br;
        acc[g] = WMMA_F16(a, b, acc[g]);
      }
    }
    float hnew[8];
#pragma unroll
    for (int r = 0; r < 8; ++r) {
      int  brow = r + hi * 8;                       // 0..15
      long zrow = (long)(b0 + brow) * T + t;
      const _Float16* zp = XZ + zrow * (4 * U);
      float zi = acc[0][r] + (float)zp[0 * U + ncol];
      float zf = acc[1][r] + (float)zp[1 * U + ncol];
      float zg = acc[2][r] + (float)zp[2 * U + ncol];
      float zo = acc[3][r] + (float)zp[3 * U + ncol];
      float ig = 1.f / (1.f + __expf(-zi));
      float fg = 1.f / (1.f + __expf(-zf));
      float og = 1.f / (1.f + __expf(-zo));
      float cv = fg * c[r] + ig * fmaxf(zg, 0.f);
      c[r] = cv;
      float hv = og * fmaxf(cv, 0.f);
      hnew[r] = hv;
      Hseq[zrow * U + ncol] = (_Float16)hv;
    }
    __syncthreads();                                // everyone done reading old h
#pragma unroll
    for (int r = 0; r < 8; ++r)
      hbuf[(r + hi * 8) * LSTR + ncol] = (_Float16)hnew[r];
    __syncthreads();                                // new h visible
  }
}

// ---------------------------------------------------------------------------
// Dense head: relu(h@Wd1+b) -> relu(@Wd2+b) -> softmax(@Wd3+b). One wave per
// batch sample (8 samples / 256-thread block). Work is ~19 MFLOP total.
// ---------------------------------------------------------------------------
__global__ __launch_bounds__(256)
void k_head(const _Float16* __restrict__ H3,
            const float* __restrict__ Wd1, const float* __restrict__ bd1,
            const float* __restrict__ Wd2, const float* __restrict__ bd2,
            const float* __restrict__ Wd3, const float* __restrict__ bd3,
            float* __restrict__ out, int T) {
  __shared__ float sh[8][64], sd1[8][64], sd2[8][32], sl[8][104];
  const int lane = threadIdx.x & 31;
  const int ww   = threadIdx.x >> 5;
  const int b    = blockIdx.x * 8 + ww;

  const _Float16* h = H3 + ((long)b * T + (T - 1)) * 64;
  sh[ww][lane]      = (float)h[lane];
  sh[ww][lane + 32] = (float)h[lane + 32];
  __syncthreads();
#pragma unroll
  for (int jj = 0; jj < 2; ++jj) {
    int j = lane + jj * 32;
    float s = bd1[j];
    for (int k = 0; k < 64; ++k) s += sh[ww][k] * Wd1[k * 64 + j];
    sd1[ww][j] = fmaxf(s, 0.f);
  }
  __syncthreads();
  {
    float s = bd2[lane];
    for (int k = 0; k < 64; ++k) s += sd1[ww][k] * Wd2[k * 32 + lane];
    sd2[ww][lane] = fmaxf(s, 0.f);
  }
  __syncthreads();
  for (int j = lane; j < 101; j += 32) {
    float s = bd3[j];
    for (int k = 0; k < 32; ++k) s += sd2[ww][k] * Wd3[k * 101 + j];
    sl[ww][j] = s;
  }
  __syncthreads();
  if (lane == 0) {
    float m = -1e30f;
    for (int j = 0; j < 101; ++j) m = fmaxf(m, sl[ww][j]);
    float s = 0.f;
    for (int j = 0; j < 101; ++j) s += __expf(sl[ww][j] - m);
    sl[ww][101] = m;
    sl[ww][102] = 1.f / s;
  }
  __syncthreads();
  float m = sl[ww][101], inv = sl[ww][102];
  for (int j = lane; j < 101; j += 32)
    out[(long)b * 101 + j] = __expf(sl[ww][j] - m) * inv;
}

// ---------------------------------------------------------------------------
extern "C" void kernel_launch(void* const* d_in, const int* in_sizes, int n_in,
                              void* d_out, int out_size, void* d_ws, size_t ws_size,
                              hipStream_t stream) {
  (void)in_sizes; (void)n_in; (void)out_size; (void)ws_size;
  constexpr int B = 1024, T = 120, D = 132, Dp = 160;
  constexpr long M = (long)B * T;                 // 122880

  const float* x   = (const float*)d_in[0];
  const float* W1  = (const float*)d_in[1];
  const float* U1  = (const float*)d_in[2];
  const float* b1  = (const float*)d_in[3];
  const float* W2  = (const float*)d_in[4];
  const float* U2  = (const float*)d_in[5];
  const float* b2  = (const float*)d_in[6];
  const float* W3  = (const float*)d_in[7];
  const float* U3  = (const float*)d_in[8];
  const float* b3  = (const float*)d_in[9];
  const float* Wd1 = (const float*)d_in[10];
  const float* bd1 = (const float*)d_in[11];
  const float* Wd2 = (const float*)d_in[12];
  const float* bd2 = (const float*)d_in[13];
  const float* Wd3 = (const float*)d_in[14];
  const float* bd3 = (const float*)d_in[15];
  float* out = (float*)d_out;

  char* ws = (char*)d_ws;
  // workspace layout (bytes, all offsets 256-aligned)
  _Float16* x16 = (_Float16*)(ws + 0);            // M x 160          (39321600)
  _Float16* wt1 = (_Float16*)(ws + 39321600);     // 256 x 160        (81920)
  _Float16* ut1 = (_Float16*)(ws + 39403520);     // 256 x 64         (32768)
  _Float16* wt2 = (_Float16*)(ws + 39436288);     // 512 x 64         (65536)
  _Float16* ut2 = (_Float16*)(ws + 39501824);     // 512 x 128        (131072)
  _Float16* wt3 = (_Float16*)(ws + 39632896);     // 256 x 128        (65536)
  _Float16* ut3 = (_Float16*)(ws + 39698432);     // 256 x 64         (32768)
  _Float16* xz  = (_Float16*)(ws + 39731200);     // M x 512 (reused) (125829120)
  _Float16* h1  = (_Float16*)(ws + 165560320);    // M x 64           (15728640)
  _Float16* h2  = (_Float16*)(ws + 181288960);    // M x 128          (31457280)
  _Float16* h3  = (_Float16*)(ws + 212746240);    // M x 64           (15728640)

  // --- precision conversion / weight transposes -------------------------
  {
    long tot = M * Dp;
    k_cvt_pad<<<dim3((unsigned)((tot + 255) / 256)), 256, 0, stream>>>(x, x16, (int)M, D, Dp);
  }
  auto tw = [&](const float* src, _Float16* dst, int K, int N, int Kp) {
    long tot = (long)N * Kp;
    k_transpose_pad<<<dim3((unsigned)((tot + 255) / 256)), 256, 0, stream>>>(src, dst, K, N, Kp);
  };
  tw(W1, wt1, 132, 256, 160);
  tw(U1, ut1,  64, 256,  64);
  tw(W2, wt2,  64, 512,  64);
  tw(U2, ut2, 128, 512, 128);
  tw(W3, wt3, 128, 256, 128);
  tw(U3, ut3,  64, 256,  64);

  const unsigned mg = (unsigned)(M / 64);         // 1920

  // --- layer 1 ----------------------------------------------------------
  k_gemm_bias<<<dim3(mg, 2), 256, 0, stream>>>(x16, wt1, b1, xz, 160, 256);
  k_lstm_rec<64><<<dim3(B / 16), 128, 0, stream>>>(xz, ut1, h1, T);
  // --- layer 2 ----------------------------------------------------------
  k_gemm_bias<<<dim3(mg, 4), 256, 0, stream>>>(h1, wt2, b2, xz, 64, 512);
  k_lstm_rec<128><<<dim3(B / 16), 256, 0, stream>>>(xz, ut2, h2, T);
  // --- layer 3 ----------------------------------------------------------
  k_gemm_bias<<<dim3(mg, 2), 256, 0, stream>>>(h2, wt3, b3, xz, 128, 256);
  k_lstm_rec<64><<<dim3(B / 16), 128, 0, stream>>>(xz, ut3, h3, T);
  // --- dense head + softmax --------------------------------------------
  k_head<<<dim3(B / 8), 256, 0, stream>>>(h3, Wd1, bd1, Wd2, bd2, Wd3, bd3, out, T);
}
